// VarianceAdaptor_16612933501342
// MI455X (gfx1250) — matrix operand order repaired and checked by reading, not compiled, not run on hardware
//
#include <hip/hip_runtime.h>
#include <hip/hip_bf16.h>

// ---------------- problem constants ----------------
#define BB   32
#define TT   512
#define MM   2048
#define DD   384
#define FF   384
#define NBQ  256

// conv-as-GEMM constants (Cin = Cout = 384, K-taps = 3)
#define CIN   384
#define COUT  384
#define KCH   36            // (3*384)/32 K-chunks of 32
#define NT    24            // COUT/16 N-tiles
#define MTILE 32            // positions per workgroup
#define SLABROWS (MTILE + 2)
#define ROWPAD 392          // padded halves per LDS slab row (bank-conflict avoidance)
#define LNSTRIDE 388        // padded floats per LN row
#define SLAB_BYTES (SLABROWS * ROWPAD * 2)       // 26656
#define LN_BYTES   (MTILE * LNSTRIDE * 4)        // 49664
#define SMEM_BYTES LN_BYTES                      // >= SLAB_BYTES

typedef _Float16 v16h __attribute__((ext_vector_type(16)));
typedef _Float16 v8h  __attribute__((ext_vector_type(8)));
typedef float    v8f  __attribute__((ext_vector_type(8)));

static __device__ __forceinline__ v16h ld_frag(const _Float16* p) {
  v8h lo = *reinterpret_cast<const v8h*>(p);
  v8h hi = *reinterpret_cast<const v8h*>(p + 16);
  return __builtin_shufflevector(lo, hi, 0,1,2,3,4,5,6,7,8,9,10,11,12,13,14,15);
}
static __device__ __forceinline__ v16h ld_frag_packed(const _Float16* p) {
  v8h lo = *reinterpret_cast<const v8h*>(p);
  v8h hi = *reinterpret_cast<const v8h*>(p + 8);
  return __builtin_shufflevector(lo, hi, 0,1,2,3,4,5,6,7,8,9,10,11,12,13,14,15);
}

// ---------------- f32 -> f16 cast ----------------
__global__ __launch_bounds__(256) void cast_f16_kernel(const float* __restrict__ x,
                                                       _Float16* __restrict__ y, size_t n) {
  for (size_t i = (size_t)blockIdx.x * 256 + threadIdx.x; i < n; i += (size_t)gridDim.x * 256)
    y[i] = (_Float16)x[i];
}

// ---------------- weight pre-pack into WMMA B-fragment order ----------------
// w: [3, CIN, COUT] f32.  wp: [KCH][NT][lane=32][h=16] f16.
// B-matrix 32x16 f16 layout: lane l -> column N = l&15 ; half h -> K = 16*(h>=8) + 8*(l>=16) + (h&7).
__global__ __launch_bounds__(256) void pack_w_kernel(const float* __restrict__ w,
                                                     _Float16* __restrict__ wp) {
  int i = blockIdx.x * 256 + threadIdx.x;
  const int total = KCH * NT * 32 * 16;
  if (i >= total) return;
  int h    = i & 15;
  int lane = (i >> 4) & 31;
  int nt   = (i >> 9) % NT;
  int kc   = (i >> 9) / NT;
  int klocal = ((h & 8) << 1) + ((lane >= 16) ? 8 : 0) + (h & 7);
  int gk = kc * 32 + klocal;          // = kk*CIN + ci (chunks never cross taps: 384/32==12)
  int n  = nt * 16 + (lane & 15);
  wp[i] = (_Float16)w[(size_t)gk * COUT + n];
}

// ---------------- fused conv1d(K=3) + bias + ReLU + LayerNorm via WMMA ----------------
// One workgroup (8 waves) computes 32 positions x 384 channels.
// Each wave: 2 M-tiles x 3 N-tiles = 6 accumulators; each B fragment reused twice.
__global__ __launch_bounds__(256)
void conv_relu_ln_wmma_kernel(const _Float16* __restrict__ Xh,
                              const _Float16* __restrict__ Wp,
                              const float* __restrict__ bias,
                              const float* __restrict__ gamma,
                              const float* __restrict__ beta,
                              _Float16* __restrict__ Yh, int L) {
  __shared__ __align__(16) unsigned char smem_raw[SMEM_BYTES];
  _Float16* slab  = reinterpret_cast<_Float16*>(smem_raw);
  float*    lnbuf = reinterpret_cast<float*>(smem_raw);

  const int tid  = threadIdx.x;
  const int lane = tid & 31;
  const int wave = tid >> 5;
  const int nTiles = L >> 5;                 // 32 positions per block
  const int bat = blockIdx.x / nTiles;
  const int p0  = (blockIdx.x % nTiles) << 5;

  // Stage im2col halo: rows p0-1 .. p0+32 (34 rows) x 384 ch, f16, padded stride.
  for (int i = tid; i < SLABROWS * 48; i += 256) {
    int r  = i / 48;
    int c8 = (i % 48) << 3;
    int p  = p0 + r - 1;
    v8h v = {};
    if (p >= 0 && p < L)
      v = *reinterpret_cast<const v8h*>(Xh + ((size_t)bat * L + p) * CIN + c8);
    *reinterpret_cast<v8h*>(slab + r * ROWPAD + c8) = v;
  }
  __syncthreads();

  v8f acc00 = {}, acc01 = {}, acc02 = {};    // m-tile 0 (positions p0..p0+15)
  v8f acc10 = {}, acc11 = {}, acc12 = {};    // m-tile 1 (positions p0+16..p0+31)
  const int nt0  = wave * 3;
  const int asel = (lane >> 4) << 3;         // A layout: lanes>=16 start at K+8
  // A base: row M = lane&15 (+16 for m-tile 1); conv tap kk shifts slab row by kk.
  const _Float16* abase = slab + (lane & 15) * ROWPAD + asel;
  const _Float16* bp    = Wp + ((size_t)(nt0 * 32) + lane) * 16;

#pragma unroll
  for (int kk = 0; kk < 3; ++kk) {
    const _Float16* a0p = abase + kk * ROWPAD;
    const _Float16* a1p = a0p + 16 * ROWPAD;
    for (int cc = 0; cc < 12; ++cc) {
      v16h a0 = ld_frag(a0p);  a0p += 32;
      v16h a1 = ld_frag(a1p);  a1p += 32;
      v16h b0 = ld_frag_packed(bp);
      v16h b1 = ld_frag_packed(bp + 512);
      v16h b2 = ld_frag_packed(bp + 1024);
      bp += NT * 512;                        // next K-chunk
      acc00 = __builtin_amdgcn_wmma_f32_16x16x32_f16(false, a0, false, b0, (short)0, acc00, false, false);
      acc10 = __builtin_amdgcn_wmma_f32_16x16x32_f16(false, a1, false, b0, (short)0, acc10, false, false);
      acc01 = __builtin_amdgcn_wmma_f32_16x16x32_f16(false, a0, false, b1, (short)0, acc01, false, false);
      acc11 = __builtin_amdgcn_wmma_f32_16x16x32_f16(false, a1, false, b1, (short)0, acc11, false, false);
      acc02 = __builtin_amdgcn_wmma_f32_16x16x32_f16(false, a0, false, b2, (short)0, acc02, false, false);
      acc12 = __builtin_amdgcn_wmma_f32_16x16x32_f16(false, a1, false, b2, (short)0, acc12, false, false);
    }
  }
  __syncthreads();   // slab no longer needed; lnbuf aliases it

  // C layout: VGPR r, lane l -> M = r + 8*(l>=16), N = l&15. Bias + ReLU -> LDS.
  const int mbase = (lane >> 4) << 3;
  {
    int n0 = nt0 * 16 + (lane & 15);
    float bs0 = bias[n0], bs1 = bias[n0 + 16], bs2 = bias[n0 + 32];
#pragma unroll
    for (int r = 0; r < 8; ++r) {
      float* row0 = lnbuf + (mbase + r) * LNSTRIDE;
      float* row1 = row0 + 16 * LNSTRIDE;
      float v;
      v = acc00[r] + bs0; row0[n0]      = v > 0.f ? v : 0.f;
      v = acc01[r] + bs1; row0[n0 + 16] = v > 0.f ? v : 0.f;
      v = acc02[r] + bs2; row0[n0 + 32] = v > 0.f ? v : 0.f;
      v = acc10[r] + bs0; row1[n0]      = v > 0.f ? v : 0.f;
      v = acc11[r] + bs1; row1[n0 + 16] = v > 0.f ? v : 0.f;
      v = acc12[r] + bs2; row1[n0 + 32] = v > 0.f ? v : 0.f;
    }
  }
  __syncthreads();

  // LayerNorm over 384 channels per position; 16 threads per position, 2 passes.
  const int sub = tid & 15;
#pragma unroll
  for (int half = 0; half < 2; ++half) {
    const int p = (tid >> 4) + (half << 4);
    float sum = 0.f, sq = 0.f;
    float xv[24];
#pragma unroll
    for (int j = 0; j < 24; ++j) {
      float x = lnbuf[p * LNSTRIDE + sub + j * 16];
      xv[j] = x; sum += x; sq += x * x;
    }
#pragma unroll
    for (int msk = 8; msk >= 1; msk >>= 1) {
      sum += __shfl_xor(sum, msk, 32);
      sq  += __shfl_xor(sq,  msk, 32);
    }
    const float mean = sum * (1.f / 384.f);
    const float var  = sq * (1.f / 384.f) - mean * mean;
    const float rstd = rsqrtf(var + 1e-5f);
    _Float16* yr = Yh + ((size_t)bat * L + p0 + p) * COUT;
#pragma unroll
    for (int j = 0; j < 24; ++j) {
      int c = sub + j * 16;
      float y = (xv[j] - mean) * rstd * gamma[c] + beta[c];
      yr[c] = (_Float16)y;
    }
  }
}

// ---------------- linear head: y[row] = h[row,:].wl + bl ----------------
__global__ __launch_bounds__(256)
void linear_head_kernel(const _Float16* __restrict__ H, const float* __restrict__ wl,
                        const float* __restrict__ bl, float* __restrict__ out, int rows) {
  int gwarp = (blockIdx.x * 256 + threadIdx.x) >> 5;
  int lane  = threadIdx.x & 31;
  if (gwarp >= rows) return;
  const _Float16* h = H + (size_t)gwarp * FF;
  float s = 0.f;
  for (int c = lane; c < FF; c += 32) s += (float)h[c] * wl[c];
#pragma unroll
  for (int msk = 16; msk >= 1; msk >>= 1) s += __shfl_xor(s, msk, 32);
  if (lane == 0) out[gwarp] = s + bl[0];
}

// ---------------- per-row inclusive cumsum of durations ----------------
__global__ __launch_bounds__(512)
void cumsum_rows_kernel(const int* __restrict__ dur, int* __restrict__ cs) {
  __shared__ int s[TT];
  int b = blockIdx.x, t = threadIdx.x;
  int v = dur[b * TT + t]; if (v < 0) v = 0;
  s[t] = v; __syncthreads();
  for (int off = 1; off < TT; off <<= 1) {
    int x = (t >= off) ? s[t - off] : 0;
    __syncthreads();
    s[t] += x;
    __syncthreads();
  }
  cs[b * TT + t] = s[t];
}

// ---------------- length regulate: gather + f32 and f16 outputs ----------------
__global__ __launch_bounds__(128)
void length_regulate_kernel(const float* __restrict__ x, const int* __restrict__ cs,
                            float* __restrict__ outf, _Float16* __restrict__ outh) {
  int bm = blockIdx.x;
  int b = bm / MM, m = bm % MM;
  const int* c = cs + b * TT;
  int total = c[TT - 1];
  int lo = 0, hi = TT;                  // searchsorted right: first t with c[t] > m
  while (lo < hi) { int mid = (lo + hi) >> 1; if (c[mid] > m) hi = mid; else lo = mid + 1; }
  int idx = lo < (TT - 1) ? lo : (TT - 1);
  bool valid = m < total;
  const float* src = x + ((size_t)b * TT + idx) * DD;
  float* dst = outf + (size_t)bm * DD;
  _Float16* dh = outh + (size_t)bm * DD;
  for (int ch = threadIdx.x; ch < DD; ch += 128) {
    float v = valid ? src[ch] : 0.f;
    dst[ch] = v;
    dh[ch] = (_Float16)v;
  }
}

// ---------------- bucketize(linspace) + embedding add (+ optional f16 refresh) ----------------
__global__ __launch_bounds__(128)
void add_emb_kernel(const float* __restrict__ vals, const float* __restrict__ emb,
                    float* __restrict__ outf, _Float16* __restrict__ outh,
                    float qlo, float qhi) {
  int bm = blockIdx.x;
  float v = vals[bm];
  // searchsorted left over q[i] = qlo + i*(qhi-qlo)/(NBQ-2), i in [0, NBQ-1)
  int l = 0, h = NBQ - 1;
  while (l < h) {
    int mid = (l + h) >> 1;
    float q = qlo + (qhi - qlo) * (float)mid * (1.f / (float)(NBQ - 2));
    if (q < v) l = mid + 1; else h = mid;
  }
  const float* e = emb + (size_t)l * DD;
  float* dst = outf + (size_t)bm * DD;
  for (int c = threadIdx.x; c < DD; c += 128) {
    float nv = dst[c] + e[c];
    dst[c] = nv;
    if (outh) outh[(size_t)bm * DD + c] = (_Float16)nv;
  }
}

// ---------------- orchestration ----------------
extern "C" void kernel_launch(void* const* d_in, const int* in_sizes, int n_in,
                              void* d_out, int out_size, void* d_ws, size_t ws_size,
                              hipStream_t stream) {
  (void)in_sizes; (void)n_in; (void)out_size; (void)ws_size;
  const float* inputs      = (const float*)d_in[0];
  const int*   true_dur    = (const int*)d_in[1];
  const float* true_pitch  = (const float*)d_in[2];
  const float* true_energy = (const float*)d_in[3];
  // d_in[4] = mel_max_len (scalar; compile-time MM)
  // pytree flatten (alphabetical): dur{b1,b2,be1,be2,bl,g1,g2,w1,w2,wl},
  // energy{...}, energy_emb, pitch{...}, pitch_emb
  const float *du_b1=(const float*)d_in[5],  *du_b2=(const float*)d_in[6],
              *du_be1=(const float*)d_in[7], *du_be2=(const float*)d_in[8],
              *du_bl=(const float*)d_in[9],  *du_g1=(const float*)d_in[10],
              *du_g2=(const float*)d_in[11], *du_w1=(const float*)d_in[12],
              *du_w2=(const float*)d_in[13], *du_wl=(const float*)d_in[14];
  const float *en_b1=(const float*)d_in[15], *en_b2=(const float*)d_in[16],
              *en_be1=(const float*)d_in[17],*en_be2=(const float*)d_in[18],
              *en_bl=(const float*)d_in[19], *en_g1=(const float*)d_in[20],
              *en_g2=(const float*)d_in[21], *en_w1=(const float*)d_in[22],
              *en_w2=(const float*)d_in[23], *en_wl=(const float*)d_in[24];
  const float *energy_emb=(const float*)d_in[25];
  const float *pi_b1=(const float*)d_in[26], *pi_b2=(const float*)d_in[27],
              *pi_be1=(const float*)d_in[28],*pi_be2=(const float*)d_in[29],
              *pi_bl=(const float*)d_in[30], *pi_g1=(const float*)d_in[31],
              *pi_g2=(const float*)d_in[32], *pi_w1=(const float*)d_in[33],
              *pi_w2=(const float*)d_in[34], *pi_wl=(const float*)d_in[35];
  const float *pitch_emb=(const float*)d_in[36];

  float* out       = (float*)d_out;
  float* outputs   = out;                                  // [B,M,D]
  float* durations = out + (size_t)BB * MM * DD;           // [B,T]
  float* pitches   = durations + (size_t)BB * TT;          // [B,M]
  float* energies  = pitches + (size_t)BB * MM;            // [B,M]

  // workspace carve
  char* ws = (char*)d_ws;
  size_t off = 0;
  auto carve = [&](size_t bytes) -> char* {
    char* p = ws + off;
    off = (off + bytes + 255) & ~(size_t)255;
    return p;
  };
  _Float16* Ah = (_Float16*)carve((size_t)BB * MM * DD * 2);
  _Float16* H1 = (_Float16*)carve((size_t)BB * MM * FF * 2);
  _Float16* H2 = (_Float16*)carve((size_t)BB * MM * FF * 2);
  const size_t packElems = (size_t)KCH * NT * 32 * 16;      // 442368
  _Float16* Wp[6];
  for (int i = 0; i < 6; ++i) Wp[i] = (_Float16*)carve(packElems * 2);
  int* cs = (int*)carve((size_t)BB * TT * 4);

  // pack all conv weights (L2-resident WMMA fragments)
  const int packBlocks = (int)((packElems + 255) / 256);
  pack_w_kernel<<<packBlocks, 256, 0, stream>>>(du_w1, Wp[0]);
  pack_w_kernel<<<packBlocks, 256, 0, stream>>>(du_w2, Wp[1]);
  pack_w_kernel<<<packBlocks, 256, 0, stream>>>(pi_w1, Wp[2]);
  pack_w_kernel<<<packBlocks, 256, 0, stream>>>(pi_w2, Wp[3]);
  pack_w_kernel<<<packBlocks, 256, 0, stream>>>(en_w1, Wp[4]);
  pack_w_kernel<<<packBlocks, 256, 0, stream>>>(en_w2, Wp[5]);

  // ---- duration predictor on inputs (L = T) ----
  cast_f16_kernel<<<4096, 256, 0, stream>>>(inputs, Ah, (size_t)BB * TT * DD);
  conv_relu_ln_wmma_kernel<<<BB * (TT / MTILE), 256, 0, stream>>>(Ah, Wp[0], du_b1, du_g1, du_be1, H1, TT);
  conv_relu_ln_wmma_kernel<<<BB * (TT / MTILE), 256, 0, stream>>>(H1, Wp[1], du_b2, du_g2, du_be2, H2, TT);
  linear_head_kernel<<<(BB * TT) / 8, 256, 0, stream>>>(H2, du_wl, du_bl, durations, BB * TT);

  // ---- length regulate ----
  cumsum_rows_kernel<<<BB, TT, 0, stream>>>(true_dur, cs);
  length_regulate_kernel<<<BB * MM, 128, 0, stream>>>(inputs, cs, outputs, Ah);

  // ---- pitch predictor on regulated outputs (L = M) ----
  conv_relu_ln_wmma_kernel<<<BB * (MM / MTILE), 256, 0, stream>>>(Ah, Wp[2], pi_b1, pi_g1, pi_be1, H1, MM);
  conv_relu_ln_wmma_kernel<<<BB * (MM / MTILE), 256, 0, stream>>>(H1, Wp[3], pi_b2, pi_g2, pi_be2, H2, MM);
  linear_head_kernel<<<(BB * MM) / 8, 256, 0, stream>>>(H2, pi_wl, pi_bl, pitches, BB * MM);

  // outputs += pitch_emb[bucketize(true_pitch)], refresh f16 activations
  add_emb_kernel<<<BB * MM, 128, 0, stream>>>(true_pitch, pitch_emb, outputs, Ah, -3.f, 200.f);

  // ---- energy predictor ----
  conv_relu_ln_wmma_kernel<<<BB * (MM / MTILE), 256, 0, stream>>>(Ah, Wp[4], en_b1, en_g1, en_be1, H1, MM);
  conv_relu_ln_wmma_kernel<<<BB * (MM / MTILE), 256, 0, stream>>>(H1, Wp[5], en_b2, en_g2, en_be2, H2, MM);
  linear_head_kernel<<<(BB * MM) / 8, 256, 0, stream>>>(H2, en_wl, en_bl, energies, BB * MM);

  // outputs += energy_emb[bucketize(true_energy)]  (no further f16 refresh needed)
  add_emb_kernel<<<BB * MM, 128, 0, stream>>>(true_energy, energy_emb, outputs, (_Float16*)nullptr, -2.f, 200.f);
}
